// RecurrentPPOAgent_72834055406457
// MI455X (gfx1250) — matrix operand order repaired and checked by reading, not compiled, not run on hardware
//
#include <hip/hip_runtime.h>
#include <hip/hip_bf16.h>

// ---------------------------------------------------------------------------
// RecurrentPPOAgent forward for MI455X (gfx1250), wave32 + WMMA bf16.
//
//   prep_weights_kernel : one-time f32 -> bf16 fragment-swizzle of W1/W2
//                         (both branches) into workspace.
//   encoder_kernel      : x2[branch][TB][64] (bf16, in d_ws)
//                         = relu(relu(obs@W1+b1)@W2+b2); per-wave 16-row tile.
//   lstm_kernel         : per (branch, 16-row batch tile) workgroup iterates
//                         T=512 steps; gates via v_wmma_f32_16x16x32_bf16;
//                         logits/value heads fused into the loop.
//
// All LDS fragment traffic is 16-byte vectorized (ds_load_b128/ds_store_b64).
// Workspace: 128 MiB (x2 bf16, both branches) + 48 KiB (swizzled weights).
//
// d_in (setup_inputs() dict order, params flattened in insertion order):
//   0 obs[T*B*128] 1 dones[T*B]
//   2..9   actor_{W1,b1,W2,b2,Wih,Whh,bih,bhh}
//   10..17 critic_{W1,b1,W2,b2,Wih,Whh,bih,bhh}
//   18 logits_W[64*16] 19 logits_b[16] 20 value_W[64] 21 value_b[1]
//   22 actor_h 23 actor_c 24 critic_h 25 critic_c   (each [B*64])
// d_out: logits[T*B*16] | values[T*B] | ah | ac | ch | cc  (f32)
// ---------------------------------------------------------------------------

#define T_DIM 512
#define B_DIM 1024
#define OBS_DIM 128
#define H_DIM 64
#define G_DIM 256
#define A_DIM 16
#define TB_DIM (T_DIM * B_DIM)

#define W1F_ELEMS (16 * 32 * 16)  // 8192 u16, 16 KiB
#define W2F_ELEMS (8 * 32 * 16)   // 4096 u16, 8 KiB
#define WPREP_PER_BRANCH (W1F_ELEMS + W2F_ELEMS)
#define X2_ELEMS ((size_t)2 * TB_DIM * H_DIM)  // u16 elements

typedef __attribute__((ext_vector_type(16))) __bf16 v16bf;
typedef __attribute__((ext_vector_type(8))) float v8f;
typedef __attribute__((ext_vector_type(4))) float v4f;
typedef __attribute__((ext_vector_type(4))) unsigned int v4u;
typedef __attribute__((ext_vector_type(2))) unsigned int v2u;

union Frag {
    v16bf v;
    unsigned short u[16];
    v4u q[2];
};

__device__ inline unsigned short f2bf(float f) {
    unsigned int x = __float_as_uint(f);
    x += 0x7fffu + ((x >> 16) & 1u);  // round to nearest even
    return (unsigned short)(x >> 16);
}
__device__ inline float bf2f(unsigned int u) {
    return __uint_as_float(u << 16);
}

__device__ inline v8f wmma_bf16(const Frag& a, const Frag& b, v8f c) {
    return __builtin_amdgcn_wmma_f32_16x16x32_bf16(
        false, a.v, false, b.v, (short)0, c, false, false);
}

// A-matrix fragment (16x32 bf16) from a row-major bf16 buffer (16B aligned).
// lane<16: row=lane, K = kbase+{0..15}; lane>=16: K = kbase+{16..31}.
// Two ds_load_b128 per fragment.
__device__ inline void loadA(Frag& f, const unsigned short* buf, int stride,
                             int kbase, int lane) {
    const int row = lane & 15;
    const int khalf = (lane >> 4) << 4;  // 0 or 16
    const v4u* p = (const v4u*)(buf + row * stride + kbase + khalf);
    f.q[0] = p[0];  // K + 0..7
    f.q[1] = p[1];  // K + 8..15
}

// B-matrix fragment from fragment-swizzled LDS: layout [frag][lane][16].
__device__ inline void loadBsw(Frag& f, const unsigned short* Wf, int fi,
                               int lane) {
    const v4u* p = (const v4u*)(Wf + (((fi << 5) + lane) << 4));
    f.q[0] = p[0];
    f.q[1] = p[1];
}

// B-matrix fragment (32x16 bf16) gathered from row-major f32 weights W[K][ldn]
// (one-time, loop-invariant).
__device__ inline void loadBg(Frag& f, const float* __restrict__ W, int ldn,
                              int kbase, int nbase, int lane) {
    const int n2 = lane & 15;
    const int kl = (lane >> 4) << 4;  // 0 or 16
#pragma unroll
    for (int j = 0; j < 16; ++j)
        f.u[j] = f2bf(W[(size_t)(kbase + kl + j) * ldn + nbase + n2]);
}

// ---------------------------------------------------------------------------
// One-time weight prep: fragment-swizzle W1/W2 (f32 -> bf16) into workspace.
// Layout per branch: [W1 frags: (nt*4+kb)][lane][16] then [W2 frags:
// (nt*2+kb)][lane][16].
// ---------------------------------------------------------------------------
__global__ __launch_bounds__(128) void prep_weights_kernel(
    const float* __restrict__ W1a, const float* __restrict__ W2a,
    const float* __restrict__ W1c, const float* __restrict__ W2c,
    unsigned short* __restrict__ wprep) {
    const int branch = blockIdx.x;
    const int tid = threadIdx.x;
    const float* W1 = branch ? W1c : W1a;
    const float* W2 = branch ? W2c : W2a;
    unsigned short* dst = wprep + (size_t)branch * WPREP_PER_BRANCH;

    for (int i = tid; i < W1F_ELEMS; i += 128) {
        const int j = i & 15, ln = (i >> 4) & 31, fi = i >> 9;
        const int nt = fi >> 2, kb = fi & 3;
        const int k = kb * 32 + ((ln >> 4) << 4) + j;
        const int n = nt * 16 + (ln & 15);
        dst[i] = f2bf(W1[k * H_DIM + n]);
    }
    for (int i = tid; i < W2F_ELEMS; i += 128) {
        const int j = i & 15, ln = (i >> 4) & 31, fi = i >> 9;
        const int nt = fi >> 1, kb = fi & 1;
        const int k = kb * 32 + ((ln >> 4) << 4) + j;
        const int n = nt * 16 + (ln & 15);
        dst[W1F_ELEMS + i] = f2bf(W2[k * H_DIM + n]);
    }
}

// ---------------------------------------------------------------------------
// Encoder: 4 waves/block, each wave computes one 16-row tile of x2 (bf16).
// ---------------------------------------------------------------------------
__global__ __launch_bounds__(128) void encoder_kernel(
    const float* __restrict__ obs, const float* __restrict__ b1a,
    const float* __restrict__ b2a, const float* __restrict__ b1c,
    const float* __restrict__ b2c, const unsigned short* __restrict__ wprep,
    unsigned short* __restrict__ x2ws) {
    __align__(16) __shared__ unsigned short sW1f[W1F_ELEMS];  // 16 KB
    __align__(16) __shared__ unsigned short sW2f[W2F_ELEMS];  // 8 KB
    __shared__ float sB1[H_DIM], sB2[H_DIM];
    __align__(16) __shared__ unsigned short sObs[4][16 * OBS_DIM];  // 16 KB
    __align__(16) __shared__ unsigned short sX1[4][16 * H_DIM];     // 8 KB

    const int tid = threadIdx.x, wave = tid >> 5, lane = tid & 31;
    const int branch = blockIdx.y;
    const float* B1 = branch ? b1c : b1a;
    const float* B2 = branch ? b2c : b2a;

    // Pre-swizzled bf16 weights: pure 16B copies into LDS.
    {
        const v4u* src =
            (const v4u*)(wprep + (size_t)branch * WPREP_PER_BRANCH);
#pragma unroll
        for (int i = 0; i < W1F_ELEMS / 8 / 128; ++i)
            ((v4u*)sW1f)[tid + 128 * i] = src[tid + 128 * i];
        src += W1F_ELEMS / 8;
#pragma unroll
        for (int i = 0; i < W2F_ELEMS / 8 / 128; ++i)
            ((v4u*)sW2f)[tid + 128 * i] = src[tid + 128 * i];
    }
    if (tid < H_DIM) {
        sB1[tid] = B1[tid];
        sB2[tid] = B2[tid];
    }

    const size_t r0 = ((size_t)blockIdx.x * 4 + wave) * 16;
    const float* op = obs + r0 * OBS_DIM;
    for (int i = lane; i < 16 * OBS_DIM; i += 32) sObs[wave][i] = f2bf(op[i]);
    __syncthreads();

    const int n2 = lane & 15;
    const int Mrow = (lane < 16) ? 0 : 8;

    Frag af[4];
#pragma unroll
    for (int kb = 0; kb < 4; ++kb)
        loadA(af[kb], sObs[wave], OBS_DIM, kb * 32, lane);

    // layer 1: [16,128]@[128,64] + b1, relu -> sX1
#pragma unroll
    for (int nt = 0; nt < 4; ++nt) {
        const float bv = sB1[nt * 16 + n2];
        v8f acc = {bv, bv, bv, bv, bv, bv, bv, bv};
#pragma unroll
        for (int kb = 0; kb < 4; ++kb) {
            Frag bw;
            loadBsw(bw, sW1f, nt * 4 + kb, lane);
            acc = wmma_bf16(af[kb], bw, acc);
        }
#pragma unroll
        for (int v = 0; v < 8; ++v)
            sX1[wave][(Mrow + v) * H_DIM + nt * 16 + n2] =
                f2bf(fmaxf(acc[v], 0.f));
    }

    Frag a2[2];
#pragma unroll
    for (int kb = 0; kb < 2; ++kb) loadA(a2[kb], sX1[wave], H_DIM, kb * 32, lane);

    // layer 2: [16,64]@[64,64] + b2, relu -> sX1 (reuse) -> coalesced global.
#pragma unroll
    for (int nt = 0; nt < 4; ++nt) {
        const float bv = sB2[nt * 16 + n2];
        v8f acc = {bv, bv, bv, bv, bv, bv, bv, bv};
#pragma unroll
        for (int kb = 0; kb < 2; ++kb) {
            Frag bw;
            loadBsw(bw, sW2f, nt * 2 + kb, lane);
            acc = wmma_bf16(a2[kb], bw, acc);
        }
#pragma unroll
        for (int v = 0; v < 8; ++v)
            sX1[wave][(Mrow + v) * H_DIM + nt * 16 + n2] =
                f2bf(fmaxf(acc[v], 0.f));
    }

    unsigned short* outp = x2ws + (size_t)branch * TB_DIM * H_DIM + r0 * H_DIM;
    const v4u* sp = (const v4u*)sX1[wave];
    v4u* gp = (v4u*)outp;
#pragma unroll
    for (int i = 0; i < 4; ++i) gp[lane + 32 * i] = sp[lane + 32 * i];
}

// ---------------------------------------------------------------------------
// Masked LSTM + heads. One workgroup (8 waves) per (branch, 16 batch rows),
// iterating all T=512 steps with gate weights resident in registers.
// ---------------------------------------------------------------------------
__global__ __launch_bounds__(256) void lstm_kernel(
    const unsigned short* __restrict__ x2ws, const float* __restrict__ dones,
    const float* __restrict__ Wih_a, const float* __restrict__ Whh_a,
    const float* __restrict__ bih_a, const float* __restrict__ bhh_a,
    const float* __restrict__ Wih_c, const float* __restrict__ Whh_c,
    const float* __restrict__ bih_c, const float* __restrict__ bhh_c,
    const float* __restrict__ logitsW, const float* __restrict__ logitsB,
    const float* __restrict__ valueW, const float* __restrict__ valueB,
    const float* __restrict__ h0a, const float* __restrict__ c0a,
    const float* __restrict__ h0c, const float* __restrict__ c0c,
    float* __restrict__ out) {
    __align__(16) __shared__ float gates[16 * G_DIM];         // 16 KB
    __align__(16) __shared__ unsigned short hA[16 * H_DIM];   // unmasked h_t
    __align__(16) __shared__ unsigned short hmA[16 * H_DIM];  // masked h
    __align__(16) __shared__ unsigned short xA[16 * H_DIM];   // x_t tile

    const int tid = threadIdx.x, wave = tid >> 5, lane = tid & 31;
    const int branch = blockIdx.y;
    const int rowbase = blockIdx.x * 16;
    const int n2 = lane & 15;
    const int Mrow = (lane < 16) ? 0 : 8;

    const float* Wih = branch ? Wih_c : Wih_a;
    const float* Whh = branch ? Whh_c : Whh_a;
    const float* bih = branch ? bih_c : bih_a;
    const float* bhh = branch ? bhh_c : bhh_a;
    const float* h0 = branch ? h0c : h0a;
    const float* c0 = branch ? c0c : c0a;
    const unsigned short* x2 = x2ws + (size_t)branch * TB_DIM * H_DIM;

    // Gate weight fragments stay in registers for all 512 steps.
    Frag wI[2][2], wH[2][2];
    float bsum[2];
#pragma unroll
    for (int p = 0; p < 2; ++p) {
        const int nt = wave * 2 + p;
        bsum[p] = bih[nt * 16 + n2] + bhh[nt * 16 + n2];
#pragma unroll
        for (int kb = 0; kb < 2; ++kb) {
            loadBg(wI[p][kb], Wih, G_DIM, kb * 32, nt * 16, lane);
            loadBg(wH[p][kb], Whh, G_DIM, kb * 32, nt * 16, lane);
        }
    }

    // Head weights (wave 0): actor -> logits [64,16]; critic -> value [64,1]
    // padded to an N=16 fragment with zeros.
    Frag hw[2];
    float hbias = 0.f;
    if (wave == 0) {
        if (branch == 0) {
#pragma unroll
            for (int kb = 0; kb < 2; ++kb)
                loadBg(hw[kb], logitsW, A_DIM, kb * 32, 0, lane);
            hbias = logitsB[n2];
        } else {
            const int kl = (lane >> 4) << 4;
#pragma unroll
            for (int kb = 0; kb < 2; ++kb)
#pragma unroll
                for (int j = 0; j < 16; ++j)
                    hw[kb].u[j] = (n2 == 0) ? f2bf(valueW[kb * 32 + kl + j])
                                            : (unsigned short)0;
            hbias = (n2 == 0) ? valueB[0] : 0.f;
        }
    }

    // Per-thread cell state: 4 consecutive h-columns of one row.
    const int row = tid >> 4;
    const int col0 = (tid & 15) * 4;
    const int hoff = row * H_DIM + col0;  // 8B-aligned in u16 units
    float c[4];
    const size_t srow = (size_t)(rowbase + row) * H_DIM + col0;
    {
        v2u hp;
        hp.x = (unsigned)f2bf(h0[srow + 0]) | ((unsigned)f2bf(h0[srow + 1]) << 16);
        hp.y = (unsigned)f2bf(h0[srow + 2]) | ((unsigned)f2bf(h0[srow + 3]) << 16);
        *(v2u*)(hA + hoff) = hp;
#pragma unroll
        for (int j = 0; j < 4; ++j) c[j] = c0[srow + j];
    }
    __syncthreads();

    float* outLogits = out;
    float* outValues = out + (size_t)TB_DIM * A_DIM;
    float* outStates = outValues + (size_t)TB_DIM;

    for (int t = 0; t < T_DIM; ++t) {
        // ---- Phase A: reset-mask state, stage x_t tile ----
        const float m = 1.0f - dones[(size_t)t * B_DIM + rowbase + row];
        {
            const v2u hp = *(const v2u*)(hA + hoff);
            v2u ho;
            ho.x = (unsigned)f2bf(bf2f(hp.x & 0xffffu) * m) |
                   ((unsigned)f2bf(bf2f(hp.x >> 16) * m) << 16);
            ho.y = (unsigned)f2bf(bf2f(hp.y & 0xffffu) * m) |
                   ((unsigned)f2bf(bf2f(hp.y >> 16) * m) << 16);
            *(v2u*)(hmA + hoff) = ho;
#pragma unroll
            for (int j = 0; j < 4; ++j) c[j] *= m;
        }
        {
            const v4u* src =
                (const v4u*)(x2 + (size_t)(t * B_DIM + rowbase) * H_DIM);
            ((v4u*)xA)[tid] = src[tid];
            if (t + 1 < T_DIM)
                __builtin_prefetch(
                    x2 + (size_t)((t + 1) * B_DIM + rowbase) * H_DIM + tid * 8,
                    0, 0);
        }
        __syncthreads();

        // ---- Phase B: gates = x@Wih + h*m@Whh + b via WMMA ----
        Frag xf[2], hf[2];
#pragma unroll
        for (int kb = 0; kb < 2; ++kb) {
            loadA(xf[kb], xA, H_DIM, kb * 32, lane);
            loadA(hf[kb], hmA, H_DIM, kb * 32, lane);
        }
#pragma unroll
        for (int p = 0; p < 2; ++p) {
            const int nt = wave * 2 + p;
            const float bv = bsum[p];
            v8f acc = {bv, bv, bv, bv, bv, bv, bv, bv};
#pragma unroll
            for (int kb = 0; kb < 2; ++kb) {
                acc = wmma_bf16(xf[kb], wI[p][kb], acc);
                acc = wmma_bf16(hf[kb], wH[p][kb], acc);
            }
#pragma unroll
            for (int v = 0; v < 8; ++v)
                gates[(Mrow + v) * G_DIM + nt * 16 + n2] = acc[v];
        }
        // Wave 0 overlaps the head GEMM for step t-1 (h_{t-1} = unmasked hA).
        if (wave == 0 && t > 0) {
            Frag ha[2];
#pragma unroll
            for (int kb = 0; kb < 2; ++kb) loadA(ha[kb], hA, H_DIM, kb * 32, lane);
            v8f acc = {hbias, hbias, hbias, hbias, hbias, hbias, hbias, hbias};
#pragma unroll
            for (int kb = 0; kb < 2; ++kb) acc = wmma_bf16(ha[kb], hw[kb], acc);
            if (branch == 0) {
#pragma unroll
                for (int v = 0; v < 8; ++v)
                    outLogits[((size_t)(t - 1) * B_DIM + rowbase + Mrow + v) *
                                  A_DIM +
                              n2] = acc[v];
            } else if (n2 == 0) {
#pragma unroll
                for (int v = 0; v < 8; ++v)
                    outValues[(size_t)(t - 1) * B_DIM + rowbase + Mrow + v] =
                        acc[v];
            }
        }
        __syncthreads();

        // ---- Phase C: activations + state update (vectorized LDS reads) ----
        {
            const float* gr = gates + row * G_DIM + col0;
            const v4f iv4 = *(const v4f*)(gr);
            const v4f fv4 = *(const v4f*)(gr + 64);
            const v4f gv4 = *(const v4f*)(gr + 128);
            const v4f ov4 = *(const v4f*)(gr + 192);
            float hv[4];
#pragma unroll
            for (int j = 0; j < 4; ++j) {
                const float iv = 1.f / (1.f + __expf(-iv4[j]));
                const float fv = 1.f / (1.f + __expf(-fv4[j]));
                const float gv = tanhf(gv4[j]);
                const float ov = 1.f / (1.f + __expf(-ov4[j]));
                c[j] = fv * c[j] + iv * gv;
                hv[j] = ov * tanhf(c[j]);
            }
            v2u hp;
            hp.x = (unsigned)f2bf(hv[0]) | ((unsigned)f2bf(hv[1]) << 16);
            hp.y = (unsigned)f2bf(hv[2]) | ((unsigned)f2bf(hv[3]) << 16);
            *(v2u*)(hA + hoff) = hp;
            if (t == T_DIM - 1) {
                const size_t so = (size_t)branch * 2 * B_DIM * H_DIM;
#pragma unroll
                for (int j = 0; j < 4; ++j) {
                    outStates[so + srow + j] = hv[j];
                    outStates[so + (size_t)B_DIM * H_DIM + srow + j] = c[j];
                }
            }
        }
        __syncthreads();
    }

    // Drain: head output for the final step.
    if (wave == 0) {
        Frag ha[2];
#pragma unroll
        for (int kb = 0; kb < 2; ++kb) loadA(ha[kb], hA, H_DIM, kb * 32, lane);
        v8f acc = {hbias, hbias, hbias, hbias, hbias, hbias, hbias, hbias};
#pragma unroll
        for (int kb = 0; kb < 2; ++kb) acc = wmma_bf16(ha[kb], hw[kb], acc);
        if (branch == 0) {
#pragma unroll
            for (int v = 0; v < 8; ++v)
                outLogits[((size_t)(T_DIM - 1) * B_DIM + rowbase + Mrow + v) *
                              A_DIM +
                          n2] = acc[v];
        } else if (n2 == 0) {
#pragma unroll
            for (int v = 0; v < 8; ++v)
                outValues[(size_t)(T_DIM - 1) * B_DIM + rowbase + Mrow + v] =
                    acc[v];
        }
    }
}

extern "C" void kernel_launch(void* const* d_in, const int* in_sizes, int n_in,
                              void* d_out, int out_size, void* d_ws,
                              size_t ws_size, hipStream_t stream) {
    (void)in_sizes;
    (void)n_in;
    (void)out_size;
    (void)ws_size;

    const float* obs = (const float*)d_in[0];
    const float* dones = (const float*)d_in[1];
    const float* aW1 = (const float*)d_in[2];
    const float* ab1 = (const float*)d_in[3];
    const float* aW2 = (const float*)d_in[4];
    const float* ab2 = (const float*)d_in[5];
    const float* aWih = (const float*)d_in[6];
    const float* aWhh = (const float*)d_in[7];
    const float* abih = (const float*)d_in[8];
    const float* abhh = (const float*)d_in[9];
    const float* cW1 = (const float*)d_in[10];
    const float* cb1 = (const float*)d_in[11];
    const float* cW2 = (const float*)d_in[12];
    const float* cb2 = (const float*)d_in[13];
    const float* cWih = (const float*)d_in[14];
    const float* cWhh = (const float*)d_in[15];
    const float* cbih = (const float*)d_in[16];
    const float* cbhh = (const float*)d_in[17];
    const float* lW = (const float*)d_in[18];
    const float* lB = (const float*)d_in[19];
    const float* vW = (const float*)d_in[20];
    const float* vB = (const float*)d_in[21];
    const float* h0a = (const float*)d_in[22];
    const float* c0a = (const float*)d_in[23];
    const float* h0c = (const float*)d_in[24];
    const float* c0c = (const float*)d_in[25];

    float* out = (float*)d_out;
    unsigned short* x2ws = (unsigned short*)d_ws;        // 128 MiB
    unsigned short* wprep = x2ws + X2_ELEMS;             // +48 KiB

    prep_weights_kernel<<<dim3(2), 128, 0, stream>>>(aW1, aW2, cW1, cW2,
                                                     wprep);

    dim3 egrid(TB_DIM / 16 / 4, 2);
    encoder_kernel<<<egrid, 128, 0, stream>>>(obs, ab1, ab2, cb1, cb2, wprep,
                                              x2ws);

    dim3 lgrid(B_DIM / 16, 2);
    lstm_kernel<<<lgrid, 256, 0, stream>>>(
        x2ws, dones, aWih, aWhh, abih, abhh, cWih, cWhh, cbih, cbhh, lW, lB,
        vW, vB, h0a, c0a, h0c, c0c, out);
}
